// GAT_82197084111193
// MI455X (gfx1250) — matrix operand order repaired and checked by reading, not compiled, not run on hardware
//
#include <hip/hip_runtime.h>

#define N_NODES 50000
#define N_EDGES 800000
#define F_IN    128
#define HID     64
#define HEADS   4
#define FTOT    256      // HEADS*HID
#define G_GRAPHS 256
#define C_CLASSES 10
#define LRELU_ALPHA 0.01f

typedef __attribute__((ext_vector_type(16))) _Float16 v16h;
typedef __attribute__((ext_vector_type(8)))  _Float16 v8h;
typedef __attribute__((ext_vector_type(8)))  float    v8f;

// ---------------------------------------------------------------- utilities

__global__ void fill_f32(float* __restrict__ p, float v, int n) {
    int i = blockIdx.x * blockDim.x + threadIdx.x;
    if (i < n) p[i] = v;
}

__global__ void cvt_f32_to_f16(const float* __restrict__ in, _Float16* __restrict__ out, int n) {
    int i = blockIdx.x * blockDim.x + threadIdx.x;
    if (i < n) out[i] = (_Float16)in[i];
}

// W[K x FTOT] row-major -> WT[FTOT x K] row-major, f16
__global__ void transpose_w_f16(const float* __restrict__ W, _Float16* __restrict__ WT,
                                int K, int ncol) {
    int i = blockIdx.x * blockDim.x + threadIdx.x;
    if (i >= K * ncol) return;
    int n = i / K, k = i % K;
    WT[i] = (_Float16)W[k * ncol + n];
}

// float atomic max via monotonic int trick (native global_atomic_max/min)
__device__ __forceinline__ void atomicMaxF(float* addr, float v) {
    if (v >= 0.0f) atomicMax((int*)addr, __float_as_int(v));
    else           atomicMin((unsigned int*)addr, __float_as_uint(v));
}

// ---------------------------------------------------------------- WMMA GEMM
// C[nrows x 256] = A[nrows x K](f16,row-major) * BT[256 x K](f16,row-major)^T
// One wave per 16-row strip (8 waves/block). B is shared by all 8 waves:
// staged through LDS in two halves (8 col-tiles x K each; <= 64KB static LDS),
// hot loop reads B via ds_load_b128 instead of 8x-redundant L2 traffic.
template<int K>
__global__ __launch_bounds__(256)
void gat_wmma_gemm(const _Float16* __restrict__ A, const _Float16* __restrict__ BT,
                   float* __restrict__ C, int nrows) {
    constexpr int NK = K / 32;
    constexpr int STAGE_ROWS = 128;                 // 8 col tiles of 16
    __shared__ _Float16 Bs[STAGE_ROWS * K];         // 32KB (K=128) / 64KB (K=256)

    const int lane = threadIdx.x & 31;
    const int wave = threadIdx.x >> 5;
    const int rt = blockIdx.x * 8 + wave;           // row tile index
    const bool active = (rt * 16 < nrows);
    const int mrow = lane & 15;
    const int hi   = lane >> 4;                     // lane group 0/1

    // A 16x32 f16 layout: lane m, element e -> k = hi*16 + e (contiguous)
    v16h areg[NK];
    if (active) {
        const _Float16* arow = A + (size_t)(rt * 16 + mrow) * K;
#pragma unroll
        for (int kc = 0; kc < NK; ++kc) {
            const v8h lo  = *(const v8h*)(arow + kc * 32 + hi * 16);
            const v8h hi8 = *(const v8h*)(arow + kc * 32 + hi * 16 + 8);
            v16h a;
#pragma unroll
            for (int i = 0; i < 8; ++i) { a[i] = lo[i]; a[i + 8] = hi8[i]; }
            areg[kc] = a;
        }
    }

    const int ncol = lane & 15;
    constexpr int CHUNKS = STAGE_ROWS * K / 8;      // 16B chunks per stage

#pragma unroll 1
    for (int stage = 0; stage < 2; ++stage) {
        __syncthreads();
        // cooperative stage of BT rows [stage*128, stage*128+128) into LDS
        const _Float16* bsrc = BT + (size_t)stage * STAGE_ROWS * K;
        for (int o = threadIdx.x; o < CHUNKS; o += 256)
            *(v8h*)(Bs + o * 8) = *(const v8h*)(bsrc + o * 8);
        // prefetch next stage into cache while this one computes
        if (stage == 0)
            __builtin_prefetch(BT + (size_t)STAGE_ROWS * K + (size_t)threadIdx.x * 64, 0, 1);
        __syncthreads();

        if (active) {
#pragma unroll 1
            for (int ct8 = 0; ct8 < 8; ++ct8) {
                // B 32x16 f16 layout: lane n, element e -> k = hi*16 + e (contiguous)
                const _Float16* brow = Bs + (size_t)(ct8 * 16 + ncol) * K + hi * 16;
                v8f acc = {};
#pragma unroll
                for (int kc = 0; kc < NK; ++kc) {
                    v16h b = *(const v16h*)(brow + kc * 32);
                    acc = __builtin_amdgcn_wmma_f32_16x16x32_f16(
                        false, areg[kc], false, b, (short)0, acc, false, false);
                }
                // D: element r of lane -> row = rt*16 + hi*8 + r, col
                const int ct = stage * 8 + ct8;
                float* crow = C + (size_t)(rt * 16 + hi * 8) * FTOT + ct * 16 + ncol;
#pragma unroll
                for (int r = 0; r < 8; ++r) crow[(size_t)r * FTOT] = acc[r];
            }
        }
    }
}

// ---------------------------------------------------------------- GAT pieces

// ls[n,h] = dot(Wh[n,h,:], a_src[h,:]); ld likewise
__global__ void node_attn_logits(const float* __restrict__ Wh,
                                 const float* __restrict__ a_src,
                                 const float* __restrict__ a_dst,
                                 float* __restrict__ ls, float* __restrict__ ld) {
    int i = blockIdx.x * blockDim.x + threadIdx.x;
    if (i >= N_NODES * HEADS) return;
    int n = i >> 2, h = i & 3;
    const float* w  = Wh + (size_t)n * FTOT + h * HID;
    const float* as = a_src + h * HID;
    const float* ad = a_dst + h * HID;
    float s = 0.0f, d = 0.0f;
#pragma unroll 8
    for (int f = 0; f < HID; ++f) { float x = w[f]; s += x * as[f]; d += x * ad[f]; }
    ls[i] = s; ld[i] = d;
}

__device__ __forceinline__ float edge_e(const int* src, const int* dst,
                                        const float* ls, const float* ld,
                                        int e, int h, int& d_out) {
    int s = src[e], d = dst[e];
    d_out = d;
    float v = ls[s * HEADS + h] + ld[d * HEADS + h];
    return v > 0.0f ? v : LRELU_ALPHA * v;
}

__global__ void edge_softmax_max(const int* __restrict__ src, const int* __restrict__ dst,
                                 const float* __restrict__ ls, const float* __restrict__ ld,
                                 float* __restrict__ m) {
    int i = blockIdx.x * blockDim.x + threadIdx.x;
    if (i >= N_EDGES * HEADS) return;
    int e = i >> 2, h = i & 3, d;
    float v = edge_e(src, dst, ls, ld, e, h, d);
    atomicMaxF(&m[d * HEADS + h], v);
}

__global__ void fix_max_init_z(float* __restrict__ m, float* __restrict__ z) {
    int i = blockIdx.x * blockDim.x + threadIdx.x;
    if (i >= N_NODES * HEADS) return;
    float v = m[i];
    if (!isfinite(v)) m[i] = 0.0f;
    z[i] = 0.0f;
}

__global__ void edge_softmax_exp(const int* __restrict__ src, const int* __restrict__ dst,
                                 const float* __restrict__ ls, const float* __restrict__ ld,
                                 const float* __restrict__ m,
                                 float* __restrict__ p, float* __restrict__ z) {
    int i = blockIdx.x * blockDim.x + threadIdx.x;
    if (i >= N_EDGES * HEADS) return;
    int e = i >> 2, h = i & 3, d;
    float v = edge_e(src, dst, ls, ld, e, h, d);
    float pe = __expf(v - m[d * HEADS + h]);
    p[i] = pe;
    atomicAdd(&z[d * HEADS + h], pe);
}

// one block per edge, one thread per feature channel
__global__ __launch_bounds__(256)
void edge_aggregate(const int* __restrict__ src, const int* __restrict__ dst,
                    const float* __restrict__ Wh, const float* __restrict__ p,
                    const float* __restrict__ z, float* __restrict__ agg) {
    int e = blockIdx.x;
    int t = threadIdx.x;           // 0..255 -> head = t/64, feat = t%64
    int h = t >> 6;
    int s = src[e], d = dst[e];
    float a = p[e * HEADS + h] / (z[d * HEADS + h] + 1e-16f);
    float val = Wh[(size_t)s * FTOT + t] * a;
    atomicAdd(&agg[(size_t)d * FTOT + t], val);
}

__global__ void elu_inplace(float* __restrict__ x, int n) {
    int i = blockIdx.x * blockDim.x + threadIdx.x;
    if (i < n) { float v = x[i]; x[i] = v > 0.0f ? v : (__expf(v) - 1.0f); }
}

// ---------------------------------------------------------------- pooling

__global__ void pool_scatter(const float* __restrict__ x, const int* __restrict__ batch,
                             float* __restrict__ pmax, float* __restrict__ psum) {
    int i = blockIdx.x * blockDim.x + threadIdx.x;
    if (i >= N_NODES * FTOT) return;
    int n = i >> 8, c = i & 255;
    int g = batch[n];
    float v = x[i];
    atomicMaxF(&pmax[g * FTOT + c], v);
    atomicAdd(&psum[g * FTOT + c], v);
}

__global__ void pool_count(const int* __restrict__ batch, float* __restrict__ cnt) {
    int n = blockIdx.x * blockDim.x + threadIdx.x;
    if (n < N_NODES) atomicAdd(&cnt[batch[n]], 1.0f);
}

__global__ void pool_linear(const float* __restrict__ pmax, const float* __restrict__ psum,
                            const float* __restrict__ cnt, const float* __restrict__ linW,
                            const float* __restrict__ linb, float* __restrict__ out) {
    int i = blockIdx.x * blockDim.x + threadIdx.x;
    if (i >= G_GRAPHS * C_CLASSES) return;
    int g = i / C_CLASSES, c = i % C_CLASSES;
    float acc = linb[c];
    float invc = 1.0f / fmaxf(cnt[g], 1.0f);
    for (int j = 0; j < FTOT; ++j) {
        float mx = pmax[g * FTOT + j];
        mx = isfinite(mx) ? mx : 0.0f;
        acc += mx * linW[j * C_CLASSES + c];
        acc += (psum[g * FTOT + j] * invc) * linW[(FTOT + j) * C_CLASSES + c];
    }
    out[i] = acc;
}

// ---------------------------------------------------------------- launch

extern "C" void kernel_launch(void* const* d_in, const int* in_sizes, int n_in,
                              void* d_out, int out_size, void* d_ws, size_t ws_size,
                              hipStream_t stream) {
    const float* x     = (const float*)d_in[0];
    const int*   eidx  = (const int*)d_in[1];
    const int*   batch = (const int*)d_in[2];
    const float* W1    = (const float*)d_in[3];
    const float* a1s   = (const float*)d_in[4];
    const float* a1d   = (const float*)d_in[5];
    const float* W2    = (const float*)d_in[6];
    const float* a2s   = (const float*)d_in[7];
    const float* a2d   = (const float*)d_in[8];
    const float* linW  = (const float*)d_in[9];
    const float* linb  = (const float*)d_in[10];
    const int* src = eidx;
    const int* dst = eidx + N_EDGES;
    float* out = (float*)d_out;

    // workspace carve-out
    char* ws = (char*)d_ws;
    size_t off = 0;
    auto alloc = [&](size_t bytes) -> void* {
        void* p = ws + off;
        off += (bytes + 255) & ~(size_t)255;
        return p;
    };
    _Float16* A16  = (_Float16*)alloc((size_t)N_NODES * FTOT * sizeof(_Float16));
    _Float16* WT16 = (_Float16*)alloc((size_t)FTOT * FTOT * sizeof(_Float16));
    float* Wh   = (float*)alloc((size_t)N_NODES * FTOT * sizeof(float));
    float* lsb  = (float*)alloc((size_t)N_NODES * HEADS * sizeof(float));
    float* ldb  = (float*)alloc((size_t)N_NODES * HEADS * sizeof(float));
    float* mbuf = (float*)alloc((size_t)N_NODES * HEADS * sizeof(float));
    float* zbuf = (float*)alloc((size_t)N_NODES * HEADS * sizeof(float));
    float* attn = (float*)alloc((size_t)N_EDGES * HEADS * sizeof(float));
    float* agg  = (float*)alloc((size_t)N_NODES * FTOT * sizeof(float));
    float* pmax = (float*)alloc((size_t)G_GRAPHS * FTOT * sizeof(float));
    float* psum = (float*)alloc((size_t)G_GRAPHS * FTOT * sizeof(float));
    float* cnt  = (float*)alloc((size_t)G_GRAPHS * sizeof(float));

    const float NEG_INF = -__builtin_huge_valf();
    auto blk = [](int n) { return dim3((unsigned)((n + 255) / 256)); };
    const dim3 T(256);

    const int NH = N_NODES * HEADS;
    const int EH = N_EDGES * HEADS;
    const int NF = N_NODES * FTOT;
    const int gemm_blocks = ((N_NODES + 15) / 16 + 7) / 8;

    auto run_layer = [&](const _Float16* Ain, int K, const float* W,
                         const float* as, const float* ad) {
        transpose_w_f16<<<blk(K * FTOT), T, 0, stream>>>(W, WT16, K, FTOT);
        if (K == 128)
            gat_wmma_gemm<128><<<dim3(gemm_blocks), T, 0, stream>>>(Ain, WT16, Wh, N_NODES);
        else
            gat_wmma_gemm<256><<<dim3(gemm_blocks), T, 0, stream>>>(Ain, WT16, Wh, N_NODES);
        node_attn_logits<<<blk(NH), T, 0, stream>>>(Wh, as, ad, lsb, ldb);
        fill_f32<<<blk(NH), T, 0, stream>>>(mbuf, NEG_INF, NH);
        edge_softmax_max<<<blk(EH), T, 0, stream>>>(src, dst, lsb, ldb, mbuf);
        fix_max_init_z<<<blk(NH), T, 0, stream>>>(mbuf, zbuf);
        edge_softmax_exp<<<blk(EH), T, 0, stream>>>(src, dst, lsb, ldb, mbuf, attn, zbuf);
        fill_f32<<<blk(NF), T, 0, stream>>>(agg, 0.0f, NF);
        edge_aggregate<<<dim3(N_EDGES), T, 0, stream>>>(src, dst, Wh, attn, zbuf, agg);
        elu_inplace<<<blk(NF), T, 0, stream>>>(agg, NF);
    };

    // layer 1
    cvt_f32_to_f16<<<blk(N_NODES * F_IN), T, 0, stream>>>(x, A16, N_NODES * F_IN);
    run_layer(A16, F_IN, W1, a1s, a1d);
    // layer 2
    cvt_f32_to_f16<<<blk(NF), T, 0, stream>>>(agg, A16, NF);
    run_layer(A16, FTOT, W2, a2s, a2d);

    // pooling + head
    fill_f32<<<blk(G_GRAPHS * FTOT), T, 0, stream>>>(pmax, NEG_INF, G_GRAPHS * FTOT);
    fill_f32<<<blk(G_GRAPHS * FTOT), T, 0, stream>>>(psum, 0.0f, G_GRAPHS * FTOT);
    fill_f32<<<blk(G_GRAPHS), T, 0, stream>>>(cnt, 0.0f, G_GRAPHS);
    pool_scatter<<<blk(NF), T, 0, stream>>>(agg, batch, pmax, psum);
    pool_count<<<blk(N_NODES), T, 0, stream>>>(batch, cnt);
    pool_linear<<<blk(G_GRAPHS * C_CLASSES), T, 0, stream>>>(pmax, psum, cnt, linW, linb, out);
}
